// AutoModelForRanking_78829829751029
// MI455X (gfx1250) — matrix operand order repaired and checked by reading, not compiled, not run on hardware
//
#include <hip/hip_runtime.h>
#include <hip/hip_bf16.h>

#define BB     128
#define LQ     512
#define LD     512
#define DD     128
#define QTILE  128
#define NTILE  64
#define NDT    (LD / NTILE)     // 8 doc tiles
#define QT     (LQ / QTILE)     // 4 query tiles per batch
#define NTHR   256

typedef __attribute__((ext_vector_type(16))) __bf16 v16bf;
typedef __attribute__((ext_vector_type(8)))  __bf16 bfx8;
typedef __attribute__((ext_vector_type(4)))  __bf16 bfx4;
typedef __attribute__((ext_vector_type(8)))  float  v8f;

union Frag { v16bf v; bfx8 h[2]; };

struct HiLo { __bf16 hi, lo; };

__device__ __forceinline__ HiLo split_bf16(float x) {
    HiLo r;
    r.hi = (__bf16)x;
    r.lo = (__bf16)(x - (float)r.hi);
    return r;
}

__device__ __forceinline__ v16bf load_frag(const __bf16* row, int k0, int k1) {
    Frag f;
    f.h[0] = *(const bfx8*)(row + k0);
    f.h[1] = *(const bfx8*)(row + k1);
    return f.v;
}

__global__ __launch_bounds__(NTHR) void colbert_main_kernel(
    const float* __restrict__ q,   // [B, LQ, D]
    const float* __restrict__ d,   // [B, LD, D]
    float* __restrict__ ws)        // [B*QT] partials
{
    __shared__ __bf16 qhi_s[QTILE][DD];   // 32 KB
    __shared__ __bf16 qlo_s[QTILE][DD];   // 32 KB
    __shared__ __bf16 dhi_s[NTILE][DD];   // 16 KB
    __shared__ __bf16 dlo_s[NTILE][DD];   // 16 KB
    __shared__ float  dsq_s[LD];          //  2 KB
    __shared__ float  redbuf[NTHR];       //  1 KB
    __shared__ float  wavesum_s[NTHR / 32];

    const int tid   = threadIdx.x;
    const int wave  = tid >> 5;
    const int lane  = tid & 31;
    const int laneM = lane & 15;
    const int khalf = (lane >> 4) & 1;

    const int b  = blockIdx.x >> 2;       // batch
    const int qt = blockIdx.x & 3;        // query tile within batch

    const float* Qg = q + ((size_t)b * LQ + (size_t)qt * QTILE) * DD;
    const float* Dg = d + (size_t)b * LD * DD;

    // ---- d_sq for all 512 docs (exact f32, from global) ----
    for (int k = tid; k < LD; k += NTHR) {
        const float4* dp = (const float4*)(Dg + (size_t)k * DD);
        float s = 0.f;
        #pragma unroll 8
        for (int i = 0; i < DD / 4; i++) {
            float4 v = dp[i];
            s += v.x * v.x + v.y * v.y + v.z * v.z + v.w * v.w;
        }
        dsq_s[k] = s;
    }

    // ---- Q tile -> bf16 hi/lo in LDS; accumulate exact q_sq partial ----
    float qsqp = 0.f;
    #pragma unroll
    for (int i = 0; i < (QTILE * DD / 4) / NTHR; i++) {   // 16 iters
        int idx = tid + i * NTHR;                         // 0..4095 float4s
        int row = idx >> 5;                               // 32 float4 per row
        int c4  = (idx & 31) << 2;
        float4 v = *(const float4*)(Qg + (size_t)row * DD + c4);
        qsqp += v.x * v.x + v.y * v.y + v.z * v.z + v.w * v.w;
        bfx4 hv, lv;
        HiLo e0 = split_bf16(v.x); hv[0] = e0.hi; lv[0] = e0.lo;
        HiLo e1 = split_bf16(v.y); hv[1] = e1.hi; lv[1] = e1.lo;
        HiLo e2 = split_bf16(v.z); hv[2] = e2.hi; lv[2] = e2.lo;
        HiLo e3 = split_bf16(v.w); hv[3] = e3.hi; lv[3] = e3.lo;
        *(bfx4*)&qhi_s[row][c4] = hv;
        *(bfx4*)&qlo_s[row][c4] = lv;
    }

    const int rbase = wave * 16;                 // this wave's query rows
    const __bf16* qhrow = &qhi_s[rbase + laneM][0];
    const __bf16* qlrow = &qlo_s[rbase + laneM][0];

    float rmax[8];
    #pragma unroll
    for (int j = 0; j < 8; j++) rmax[j] = -3.0e38f;

    for (int dt = 0; dt < NDT; dt++) {
        // ---- load doc tile (64 docs x 128 dims) -> bf16 hi/lo in LDS ----
        #pragma unroll
        for (int i = 0; i < (NTILE * DD / 4) / NTHR; i++) {   // 8 iters
            int idx = tid + i * NTHR;                         // 0..2047
            int row = idx >> 5;
            int c4  = (idx & 31) << 2;
            float4 v = *(const float4*)(Dg + (size_t)(dt * NTILE + row) * DD + c4);
            bfx4 hv, lv;
            HiLo e0 = split_bf16(v.x); hv[0] = e0.hi; lv[0] = e0.lo;
            HiLo e1 = split_bf16(v.y); hv[1] = e1.hi; lv[1] = e1.lo;
            HiLo e2 = split_bf16(v.z); hv[2] = e2.hi; lv[2] = e2.lo;
            HiLo e3 = split_bf16(v.w); hv[3] = e3.hi; lv[3] = e3.lo;
            *(bfx4*)&dhi_s[row][c4] = hv;
            *(bfx4*)&dlo_s[row][c4] = lv;
        }
        __syncthreads();

        // prefetch next doc tile toward caches (global_prefetch_b8)
        if (dt + 1 < NDT) {
            const float* nb = Dg + (size_t)(dt + 1) * NTILE * DD;
            __builtin_prefetch(nb + (size_t)tid * 32, 0, 1);  // one 128B line/thread
        }

        // ---- GEMM: 16 rows x 64 cols, K = 128, 3-term bf16 split ----
        v8f acc[4];
        #pragma unroll
        for (int n = 0; n < 4; n++) acc[n] = (v8f){0.f, 0.f, 0.f, 0.f, 0.f, 0.f, 0.f, 0.f};

        #pragma unroll
        for (int kc = 0; kc < 4; kc++) {
            // A (16x32 bf16): lanes 0-15 hold K=kc*32+[0..7] & +[16..23]; lanes 16-31 +8
            int ka0 = kc * 32 + khalf * 8;
            int ka1 = kc * 32 + 16 + khalf * 8;
            v16bf ah = load_frag(qhrow, ka0, ka1);
            v16bf al = load_frag(qlrow, ka0, ka1);
            // B (32x16 bf16): lanes 0-15 hold K=kc*32+[0..7] & +[16..23]; lanes 16-31 +8
            int kb0 = kc * 32 + khalf * 8;
            int kb1 = kb0 + 16;
            #pragma unroll
            for (int n = 0; n < 4; n++) {
                const __bf16* dhrow = &dhi_s[n * 16 + laneM][0];
                const __bf16* dlrow = &dlo_s[n * 16 + laneM][0];
                v16bf bh = load_frag(dhrow, kb0, kb1);
                v16bf bl = load_frag(dlrow, kb0, kb1);
                acc[n] = __builtin_amdgcn_wmma_f32_16x16x32_bf16(
                    false, ah, false, bh, (short)0, acc[n], false, false);
                acc[n] = __builtin_amdgcn_wmma_f32_16x16x32_bf16(
                    false, ah, false, bl, (short)0, acc[n], false, false);
                acc[n] = __builtin_amdgcn_wmma_f32_16x16x32_bf16(
                    false, al, false, bh, (short)0, acc[n], false, false);
            }
        }

        // ---- epilogue: score = 2*qd - d_sq; running row max ----
        #pragma unroll
        for (int n = 0; n < 4; n++) {
            float dsqv = dsq_s[dt * NTILE + n * 16 + laneM];
            #pragma unroll
            for (int j = 0; j < 8; j++) {
                float sc = 2.0f * acc[n][j] - dsqv;
                rmax[j] = fmaxf(rmax[j], sc);
            }
        }
        __syncthreads();   // safe to overwrite doc tile
    }

    // ---- reduce max across the 16 lanes of each half-wave ----
    #pragma unroll
    for (int j = 0; j < 8; j++) {
        float v = rmax[j];
        v = fmaxf(v, __shfl_xor(v, 1, 32));
        v = fmaxf(v, __shfl_xor(v, 2, 32));
        v = fmaxf(v, __shfl_xor(v, 4, 32));
        v = fmaxf(v, __shfl_xor(v, 8, 32));
        rmax[j] = v;
    }
    // lanes 0-15 now hold maxes of rows rbase+0..7; lanes 16-31 rows rbase+8..15
    float rsum = 0.f;
    #pragma unroll
    for (int j = 0; j < 8; j++) rsum += rmax[j];
    rsum += __shfl_xor(rsum, 16, 32);            // all 16 rows of this wave

    if (lane == 0) wavesum_s[wave] = rsum;
    redbuf[tid] = qsqp;
    __syncthreads();

    if (tid == 0) {
        float tot = 0.f;
        #pragma unroll
        for (int w = 0; w < NTHR / 32; w++) tot += wavesum_s[w];
        float qs = 0.f;
        for (int t = 0; t < NTHR; t++) qs += redbuf[t];
        ws[blockIdx.x] = tot - qs;               // sum_rows(max) - sum_rows(q_sq)
    }
}

__global__ void colbert_reduce_kernel(const float* __restrict__ ws,
                                      float* __restrict__ out)
{
    int b = blockIdx.x * blockDim.x + threadIdx.x;
    if (b < BB) {
        float s = 0.f;
        #pragma unroll
        for (int t = 0; t < QT; t++) s += ws[b * QT + t];
        out[b] = s;
    }
}

extern "C" void kernel_launch(void* const* d_in, const int* in_sizes, int n_in,
                              void* d_out, int out_size, void* d_ws, size_t ws_size,
                              hipStream_t stream) {
    const float* q = (const float*)d_in[0];   // [128, 512, 128] f32
    const float* d = (const float*)d_in[1];   // [128, 512, 128] f32
    float* out = (float*)d_out;               // [128] f32
    float* ws  = (float*)d_ws;                // >= B*QT floats

    colbert_main_kernel<<<BB * QT, NTHR, 0, stream>>>(q, d, ws);
    colbert_reduce_kernel<<<1, BB, 0, stream>>>(ws, out);
}
